// MedfusionAttention_32899449487837
// MI455X (gfx1250) — matrix operand order repaired and checked by reading, not compiled, not run on hardware
//
#include <hip/hip_runtime.h>
#include <hip/hip_bf16.h>
#include <math.h>

typedef __attribute__((ext_vector_type(16))) _Float16 v16h;
typedef __attribute__((ext_vector_type(8)))  _Float16 h8;
typedef __attribute__((ext_vector_type(8)))  float    v8f;

#define NUM_HEADS 8
#define CH 40
#define C_DIM 320
#define N_DIM 2304
#define B_DIM 2
#define DPAD_Q 64   // pad d=40 -> 64 (K-dim of QK^T, two k=32 WMMA steps)
#define DPAD_V 48   // pad d=40 -> 48 (output cols of P*V^T, three 16-col tiles)
#define EPSVAL 1e-5f

// Fragment bases as used in this file (wave32, f16; contiguous k order chosen here, same for A and B):
//  A 16x32 (MxK): lane row m=l&15; elems = K[2*base..2*base+15] contiguous, base=(l&16)?8:0
//  B 32x16 (KxN): lane col n=l&15; elems = K[base16..base16+15] contiguous, base16=(l&16)?16:0
//  C/D f32 16x16: lane col n=l&15; vgpr r -> row r+((l&16)?8:0)
__device__ __forceinline__ v16h ld_afrag(const _Float16* row, int abase) {
  h8 lo = *(const h8*)(row + 2 * abase);
  h8 hi = *(const h8*)(row + 2 * abase + 8);
  return __builtin_shufflevector(lo, hi, 0, 1, 2, 3, 4, 5, 6, 7,
                                 8, 9, 10, 11, 12, 13, 14, 15);
}

// ---------------- weight fp32 -> f16 (one-shot, 320x320 each) ----------------
__global__ void wcvt_kernel(const float* __restrict__ src, _Float16* __restrict__ dst, int n) {
  int i = blockIdx.x * blockDim.x + threadIdx.x;
  if (i < n) dst[i] = (_Float16)src[i];
}

// ---------------- GroupNorm: x[B,C,N] -> xn16[B,N,C] (f16, n-major) ----------------
__global__ void gn_kernel(const float* __restrict__ x, const float* __restrict__ gamma,
                          const float* __restrict__ beta, _Float16* __restrict__ xn16) {
  const int b = blockIdx.x >> 5;
  const int g = blockIdx.x & 31;
  const int cpg = C_DIM / 32;              // 10 channels per group
  const int c0 = g * cpg;
  const int count = cpg * N_DIM;           // 23040
  const float* xb = x + ((size_t)b * C_DIM + c0) * N_DIM;

  float s = 0.f, ss = 0.f;
  for (int i = threadIdx.x; i < count; i += blockDim.x) {
    float v = xb[i];
    s += v; ss += v * v;
  }
  __shared__ float red[512];
  red[threadIdx.x] = s;
  red[256 + threadIdx.x] = ss;
  __syncthreads();
  for (int off = 128; off > 0; off >>= 1) {
    if (threadIdx.x < off) {
      red[threadIdx.x]       += red[threadIdx.x + off];
      red[256 + threadIdx.x] += red[256 + threadIdx.x + off];
    }
    __syncthreads();
  }
  const float mean = red[0] / (float)count;
  const float var  = red[256] / (float)count - mean * mean;
  const float inv  = rsqrtf(var + EPSVAL);
  _Float16* ob = xn16 + (size_t)b * N_DIM * C_DIM;
  for (int i = threadIdx.x; i < count; i += blockDim.x) {
    int cl = i / N_DIM, n = i - cl * N_DIM, c = c0 + cl;
    float v = (xb[i] - mean) * inv * gamma[c] + beta[c];
    ob[(size_t)n * C_DIM + c] = (_Float16)v;       // n-major for B-operand loads
  }
}

// ------------- QKV projection: one wave per 16x16 tile, 10 WMMA k-steps -------------
__global__ void qkv_kernel(const _Float16* __restrict__ xn16,
                           const _Float16* __restrict__ Wq16, const float* __restrict__ bq,
                           const _Float16* __restrict__ Wk16, const float* __restrict__ bk,
                           const _Float16* __restrict__ Wv16, const float* __restrict__ bv,
                           _Float16* __restrict__ Qs, _Float16* __restrict__ Ks,
                           _Float16* __restrict__ Vs) {
  const int lane  = threadIdx.x & 31;
  const int o0    = blockIdx.x * 16;
  const int n0    = blockIdx.y * 16;
  const int b     = blockIdx.z / 3;
  const int which = blockIdx.z % 3;        // 0=q, 1=k, 2=v
  const _Float16* W = (which == 0) ? Wq16 : (which == 1) ? Wk16 : Wv16;
  const float* bias = (which == 0) ? bq : (which == 1) ? bk : bv;

  const int col    = lane & 15;
  const int abase  = (lane & 16) ? 8 : 0;
  const int bbase  = (lane & 16) ? 16 : 0;
  const int half8  = (lane & 16) ? 8 : 0;
  const _Float16* Wrow = W + (size_t)(o0 + (lane & 15)) * C_DIM;
  const _Float16* xrow = xn16 + ((size_t)b * N_DIM + n0 + col) * C_DIM;

  v8f acc = {};
  for (int k0 = 0; k0 < C_DIM; k0 += 32) {
    v16h a  = ld_afrag(Wrow + k0, abase);
    v16h bb = *(const v16h*)(xrow + k0 + bbase);
    acc = __builtin_amdgcn_wmma_f32_16x16x32_f16(false, a, false, bb, (short)0, acc, false, false);
  }
  const float scale = 0.3976353643835253f; // 40^(-0.25), applied to both q and k
  const int n = n0 + col;
#pragma unroll
  for (int r = 0; r < 8; ++r) {
    int o = o0 + r + half8;
    float v = acc[r] + bias[o];
    int head = o / CH;
    int dc   = o % CH;
    int bh   = b * NUM_HEADS + head;
    if (which == 0) {
      Qs[((size_t)bh * N_DIM + n) * DPAD_Q + dc] = (_Float16)(v * scale);  // [bh][n][d]
    } else if (which == 1) {
      Ks[((size_t)bh * N_DIM + n) * DPAD_Q + dc] = (_Float16)(v * scale);  // [bh][j][d]
    } else {
      Vs[((size_t)bh * DPAD_V + dc) * N_DIM + n] = (_Float16)v;            // [bh][d][j]
    }
  }
}

// ------------- Flash attention: 4 waves/block, 16 query rows per wave -------------
__global__ void attn_kernel(const _Float16* __restrict__ Qs, const _Float16* __restrict__ Ks,
                            const _Float16* __restrict__ Vs, _Float16* __restrict__ attn16) {
  const int lane  = threadIdx.x & 31;
  const int w     = threadIdx.x >> 5;      // wave in block: 0..3
  const int bh    = blockIdx.y;            // b*8+head
  const int b     = bh >> 3, head = bh & 7;
  const int n0    = (blockIdx.x * 4 + w) * 16;
  const int m     = lane & 15;
  const int col   = lane & 15;
  const int abase = (lane & 16) ? 8 : 0;
  const int bbase = (lane & 16) ? 16 : 0;
  const int half8 = (lane & 16) ? 8 : 0;

  __shared__ __align__(32) _Float16 Plds[4][16 * 32];  // per-wave P staging

  // Q A-fragments (rows n0..n0+15, K-dim 0..63), loop-invariant
  const _Float16* qrow = Qs + ((size_t)bh * N_DIM + n0 + m) * DPAD_Q;
  const v16h aq0 = ld_afrag(qrow, abase);
  const v16h aq1 = ld_afrag(qrow + 32, abase);

  float mrow[8], lrow[8];
  v8f o0v = {}, o1v = {}, o2v = {};
#pragma unroll
  for (int r = 0; r < 8; ++r) { mrow[r] = -INFINITY; lrow[r] = 0.f; }

  const _Float16* Kbh = Ks + (size_t)bh * N_DIM * DPAD_Q;
  const _Float16* Vbh = Vs + (size_t)bh * DPAD_V * N_DIM;

  for (int jt = 0; jt < N_DIM / 32; ++jt) {
    const int j0 = jt * 32;
    if (jt + 1 < N_DIM / 32) {  // software prefetch of next tile (global_prefetch_b8)
      __builtin_prefetch(Kbh + (size_t)(j0 + 32 + lane) * DPAD_Q, 0, 3);
      __builtin_prefetch(Vbh + (size_t)(lane & 15) * N_DIM + j0 + 32, 0, 3);
    }
    v8f s0 = {}, s1 = {};
    {
      const _Float16* kr = Kbh + (size_t)(j0 + col) * DPAD_Q;
      v16h bk0 = *(const v16h*)(kr + bbase);
      v16h bk1 = *(const v16h*)(kr + 32 + bbase);
      s0 = __builtin_amdgcn_wmma_f32_16x16x32_f16(false, aq0, false, bk0, (short)0, s0, false, false);
      s0 = __builtin_amdgcn_wmma_f32_16x16x32_f16(false, aq1, false, bk1, (short)0, s0, false, false);
    }
    {
      const _Float16* kr = Kbh + (size_t)(j0 + 16 + col) * DPAD_Q;
      v16h bk0 = *(const v16h*)(kr + bbase);
      v16h bk1 = *(const v16h*)(kr + 32 + bbase);
      s1 = __builtin_amdgcn_wmma_f32_16x16x32_f16(false, aq0, false, bk0, (short)0, s1, false, false);
      s1 = __builtin_amdgcn_wmma_f32_16x16x32_f16(false, aq1, false, bk1, (short)0, s1, false, false);
    }

    // online softmax: row stats via width-16 shuffles (C/D layout: half-wave <-> 8 rows)
    float rmax[8];
#pragma unroll
    for (int r = 0; r < 8; ++r) rmax[r] = fmaxf(s0[r], s1[r]);
#pragma unroll
    for (int msk = 8; msk >= 1; msk >>= 1)
#pragma unroll
      for (int r = 0; r < 8; ++r) rmax[r] = fmaxf(rmax[r], __shfl_xor(rmax[r], msk, 16));

    float p0[8], p1[8], corr[8], rs[8];
#pragma unroll
    for (int r = 0; r < 8; ++r) {
      float mnew = fmaxf(mrow[r], rmax[r]);
      corr[r] = __expf(mrow[r] - mnew);   // exp(-inf)=0 on first chunk
      mrow[r] = mnew;
      p0[r] = __expf(s0[r] - mnew);
      p1[r] = __expf(s1[r] - mnew);
      rs[r] = p0[r] + p1[r];
    }
#pragma unroll
    for (int msk = 8; msk >= 1; msk >>= 1)
#pragma unroll
      for (int r = 0; r < 8; ++r) rs[r] += __shfl_xor(rs[r], msk, 16);
#pragma unroll
    for (int r = 0; r < 8; ++r) {
      lrow[r] = lrow[r] * corr[r] + rs[r];
      o0v[r] *= corr[r]; o1v[r] *= corr[r]; o2v[r] *= corr[r];
    }

    // stage P (f16, row-major 16x32) to LDS, regather as A-fragment (2x ds b128)
#pragma unroll
    for (int r = 0; r < 8; ++r) {
      Plds[w][(r + half8) * 32 + col]      = (_Float16)p0[r];
      Plds[w][(r + half8) * 32 + 16 + col] = (_Float16)p1[r];
    }
    __syncthreads();
    v16h ap = ld_afrag(&Plds[w][m * 32], abase);
    const _Float16* vr0 = Vbh + (size_t)col * N_DIM + j0 + bbase;
    v16h bv0 = *(const v16h*)(vr0);
    v16h bv1 = *(const v16h*)(vr0 + (size_t)16 * N_DIM);
    v16h bv2 = *(const v16h*)(vr0 + (size_t)32 * N_DIM);
    o0v = __builtin_amdgcn_wmma_f32_16x16x32_f16(false, ap, false, bv0, (short)0, o0v, false, false);
    o1v = __builtin_amdgcn_wmma_f32_16x16x32_f16(false, ap, false, bv1, (short)0, o1v, false, false);
    o2v = __builtin_amdgcn_wmma_f32_16x16x32_f16(false, ap, false, bv2, (short)0, o2v, false, false);
    __syncthreads();
  }

  // normalize and scatter to attn16[b][n][head*40+dc] (f16 n-major, B-operand of out-proj)
#pragma unroll
  for (int r = 0; r < 8; ++r) {
    float invl = 1.f / lrow[r];
    int n = n0 + r + half8;
    float vals[3] = { o0v[r] * invl, o1v[r] * invl, o2v[r] * invl };
#pragma unroll
    for (int c = 0; c < 3; ++c) {
      int dc = c * 16 + col;
      if (dc < CH) {
        attn16[((size_t)b * N_DIM + n) * C_DIM + head * CH + dc] = (_Float16)vals[c];
      }
    }
  }
}

// ------------- out = x + Wo @ attn + bo -------------
__global__ void outproj_kernel(const _Float16* __restrict__ attn16,
                               const _Float16* __restrict__ Wo16, const float* __restrict__ bo,
                               const float* __restrict__ x, float* __restrict__ out) {
  const int lane  = threadIdx.x & 31;
  const int o0    = blockIdx.x * 16;
  const int n0    = blockIdx.y * 16;
  const int b     = blockIdx.z;
  const int col   = lane & 15;
  const int abase = (lane & 16) ? 8 : 0;
  const int bbase = (lane & 16) ? 16 : 0;
  const int half8 = (lane & 16) ? 8 : 0;
  const _Float16* Wrow = Wo16 + (size_t)(o0 + (lane & 15)) * C_DIM;
  const _Float16* arow = attn16 + ((size_t)b * N_DIM + n0 + col) * C_DIM;

  v8f acc = {};
  for (int k0 = 0; k0 < C_DIM; k0 += 32) {
    v16h a  = ld_afrag(Wrow + k0, abase);
    v16h bb = *(const v16h*)(arow + k0 + bbase);
    acc = __builtin_amdgcn_wmma_f32_16x16x32_f16(false, a, false, bb, (short)0, acc, false, false);
  }
  const int n = n0 + col;
#pragma unroll
  for (int r = 0; r < 8; ++r) {
    int o = o0 + r + half8;
    size_t idx = ((size_t)b * C_DIM + o) * N_DIM + n;
    out[idx] = x[idx] + acc[r] + bo[o];
  }
}

extern "C" void kernel_launch(void* const* d_in, const int* in_sizes, int n_in,
                              void* d_out, int out_size, void* d_ws, size_t ws_size,
                              hipStream_t stream) {
  const float* x     = (const float*)d_in[0];
  const float* gamma = (const float*)d_in[1];
  const float* beta  = (const float*)d_in[2];
  const float* Wq    = (const float*)d_in[3];
  const float* bq    = (const float*)d_in[4];
  const float* Wk    = (const float*)d_in[5];
  const float* bk    = (const float*)d_in[6];
  const float* Wv    = (const float*)d_in[7];
  const float* bv    = (const float*)d_in[8];
  const float* Wo    = (const float*)d_in[9];
  const float* bo    = (const float*)d_in[10];
  float* out = (float*)d_out;

  char* ws = (char*)d_ws;
  size_t off = 0;
  auto walloc = [&](size_t bytes) {
    void* p = ws + off;
    off += (bytes + 255) & ~(size_t)255;
    return p;
  };
  const size_t xnB = (size_t)B_DIM * N_DIM * C_DIM * sizeof(_Float16);               // 2.95 MB
  const size_t qB  = (size_t)B_DIM * NUM_HEADS * N_DIM * DPAD_Q * sizeof(_Float16);  // 4.72 MB
  const size_t vB  = (size_t)B_DIM * NUM_HEADS * DPAD_V * N_DIM * sizeof(_Float16);  // 3.54 MB
  const size_t wB  = (size_t)C_DIM * C_DIM * sizeof(_Float16);                       // 0.2 MB each
  _Float16* xn16   = (_Float16*)walloc(xnB);
  _Float16* Qs     = (_Float16*)walloc(qB);
  _Float16* Ks     = (_Float16*)walloc(qB);
  _Float16* Vs     = (_Float16*)walloc(vB);
  _Float16* attn16 = (_Float16*)walloc(xnB);
  _Float16* Wq16   = (_Float16*)walloc(wB);
  _Float16* Wk16   = (_Float16*)walloc(wB);
  _Float16* Wv16   = (_Float16*)walloc(wB);
  _Float16* Wo16   = (_Float16*)walloc(wB);

  // zero-fill so d-padding (40->64 / 40->48) contributes nothing to the WMMAs
  hipMemsetAsync(Qs, 0, qB, stream);
  hipMemsetAsync(Ks, 0, qB, stream);
  hipMemsetAsync(Vs, 0, vB, stream);

  const int wn = C_DIM * C_DIM;
  wcvt_kernel<<<dim3((wn + 255) / 256), dim3(256), 0, stream>>>(Wq, Wq16, wn);
  wcvt_kernel<<<dim3((wn + 255) / 256), dim3(256), 0, stream>>>(Wk, Wk16, wn);
  wcvt_kernel<<<dim3((wn + 255) / 256), dim3(256), 0, stream>>>(Wv, Wv16, wn);
  wcvt_kernel<<<dim3((wn + 255) / 256), dim3(256), 0, stream>>>(Wo, Wo16, wn);

  gn_kernel<<<dim3(B_DIM * 32), dim3(256), 0, stream>>>(x, gamma, beta, xn16);
  qkv_kernel<<<dim3(C_DIM / 16, N_DIM / 16, B_DIM * 3), dim3(32), 0, stream>>>(
      xn16, Wq16, bq, Wk16, bk, Wv16, bv, Qs, Ks, Vs);
  attn_kernel<<<dim3(N_DIM / 64, B_DIM * NUM_HEADS), dim3(128), 0, stream>>>(Qs, Ks, Vs, attn16);
  outproj_kernel<<<dim3(C_DIM / 16, N_DIM / 16, B_DIM), dim3(32), 0, stream>>>(
      attn16, Wo16, bo, x, out);
}
